// CIFARnet_25615184954205
// MI455X (gfx1250) — compile-verified, matrix operand order checked by reading, not run on hardware
//
#include <hip/hip_runtime.h>

#define TDIV(a,b) (((a)+(b)-1)/(b))

typedef __attribute__((ext_vector_type(16))) _Float16 v16h;
typedef __attribute__((ext_vector_type(8)))  _Float16 v8h;
typedef __attribute__((ext_vector_type(2)))  _Float16 v2h;
typedef __attribute__((ext_vector_type(8)))  float    v8f;

#define DM_F 0.951229424500714f   // exp(-1/20)
#define DS_F 0.818730753077982f   // exp(-1/5)
#define NSTEPS 8.0f

// LDS row stride (halves): 40 halves = 80B -> 16B-aligned rows, bank stride
// 20 mod 64 => conflict-free across 16 lanes.
#define LROW 40

// ---------------------------------------------------------------- utilities
__device__ __forceinline__ void blockAtomicSum(float v, float* slot) {
  __shared__ float sm[256];
  const int tid = threadIdx.x;
  sm[tid] = v;
  __syncthreads();
#pragma unroll
  for (int s = 128; s > 0; s >>= 1) {
    if (tid < s) sm[tid] += sm[tid + s];
    __syncthreads();
  }
  if (tid == 0) atomicAdd(slot, sm[0]);
}

__global__ void k_fill(float* __restrict__ p, int n, float v) {
  int i = blockIdx.x * 256 + threadIdx.x;
  if (i < n) p[i] = v;
}

// one-time weight conversion f32 -> f16 (amortized across all 8 time steps)
__global__ void k_f32tof16(const float* __restrict__ src, _Float16* __restrict__ dst,
                           int n) {
  int i = blockIdx.x * 256 + threadIdx.x;
  if (i < n) dst[i] = (_Float16)src[i];
}

// ------------------------------------------------------- input encoding neuron
__global__ void k_input_encode(const float* __restrict__ x, float* __restrict__ in_u,
                               float* __restrict__ in_sp, float* __restrict__ sumslot,
                               int n) {
  int i = blockIdx.x * 256 + threadIdx.x;
  float s = 0.f;
  if (i < n) {
    float u = DM_F * in_u[i] + x[i] * 0.2f;
    s = (u - 1.0f > 0.f) ? 1.f : 0.f;
    in_u[i] = u * (1.f - s);
    in_sp[i] = s;
  }
  blockAtomicSum(s, sumslot);
}

// --------------------- dual-exponential trace + f16 side buffers for the GEMM
// conv(vm,w)-conv(vs,w) == conv(vm-vs,w) (linearity), so the GEMM consumes
// dh = f16(vm-vs); mh = f16(vm) is only read by NEED_UM layers.
__global__ void k_trace(const float* __restrict__ sp, float* __restrict__ vm,
                        float* __restrict__ vs, _Float16* __restrict__ dh,
                        _Float16* __restrict__ mh, int n) {
  int i = blockIdx.x * 256 + threadIdx.x;
  if (i < n) {
    float s = sp[i];
    float m = DM_F * vm[i] + s;
    float v = DS_F * vs[i] + s;
    vm[i] = m;
    vs[i] = v;
    dh[i] = (_Float16)(m - v);
    mh[i] = (_Float16)m;
  }
}

// ---------------------------------------------------------------------------
// Implicit-GEMM NOSO conv/FC, WMMA f16 -> f32, LDS ping-pong double buffering.
//   M = OC, N = B*H*W, K = IC*KS*KS
// 256 threads = 8 waves; 64(M) x 32(N) tile, one 16x16 WMMA tile per wave.
// Chunk i+1 is staged (async global->LDS B128 for weights + f16 im2col gather
// for activations) into the alternate LDS buffer while chunk i is computed:
// one workgroup barrier per K-chunk, ASYNCcnt drained just before it.
// ---------------------------------------------------------------------------
template <bool NEED_UM>
__global__ __launch_bounds__(256) void k_noso_gemm(
    const _Float16* __restrict__ Wh, const _Float16* __restrict__ dh,
    const _Float16* __restrict__ vmh, float* __restrict__ udiff,
    float* __restrict__ um_out, int OC, int IC, int KS, int PAD, int H, int Wd,
    int Bsz) {
  __shared__ _Float16 As[2][64 * LROW];   // [m][k]
  __shared__ _Float16 Bs[2][32 * LROW];   // [n][k] (transposed for vector reads)
  __shared__ _Float16 Bs2[2][32 * LROW];

  const int tid  = threadIdx.x;
  const int lane = tid & 31;
  const int wave = tid >> 5;
  const int mi = wave & 3;   // 4 M-subtiles
  const int ni = wave >> 2;  // 2 N-subtiles
  const int n0 = blockIdx.x * 32;
  const int m0 = blockIdx.y * 64;
  const int KK = KS * KS;
  const int Ktot = IC * KK;
  const int HW = H * Wd;
  const int Ntot = Bsz * HW;

  v8f accD = {};
  v8f accM = {};

  // per-thread coordinates for the async A-tile copy: 64 rows x 64B, one
  // 16B chunk per thread (256 threads cover the tile exactly)
  const int arow = tid >> 2;
  const int achk = tid & 3;

  auto stage = [&](int k0, int buf) {
    _Float16* Ab  = As[buf];
    _Float16* Bb  = Bs[buf];
    _Float16* B2b = Bs2[buf];
    const bool interior = (m0 + 64 <= OC) && (k0 + 32 <= Ktot);
    if (interior) {
      // CDNA5 async copy global->LDS (ASYNCcnt-tracked, no VGPR round-trip)
      const _Float16* gp = Wh + (size_t)(m0 + arow) * Ktot + k0 + achk * 8;
      const unsigned lds =
          (unsigned)(unsigned long long)(void*)(Ab + arow * LROW + achk * 8);
      asm volatile("global_load_async_to_lds_b128 %0, %1, off"
                   :: "v"(lds), "v"((unsigned long long)gp)
                   : "memory");
    } else {
      for (int i = tid; i < 64 * 16; i += 256) {
        int mm = i >> 4, kp = (i & 15) << 1;
        int m = m0 + mm, k = k0 + kp;
        v2h w;
        w[0] = (_Float16)0.f;
        w[1] = (_Float16)0.f;
        if (m < OC) {
          if (k < Ktot)     w[0] = Wh[(size_t)m * Ktot + k];
          if (k + 1 < Ktot) w[1] = Wh[(size_t)m * Ktot + k + 1];
        }
        *(v2h*)(Ab + mm * LROW + kp) = w;
      }
    }
    // prefetch the chunk after this one toward L2 (global_prefetch_b8)
    if (k0 + 32 < Ktot) {
      int m = m0 + (tid >> 2);
      if (m < OC)
        __builtin_prefetch(Wh + (size_t)m * Ktot + k0 + 32 + ((tid & 3) << 3), 0, 1);
    }
    // B tile: im2col gather of f16 traces, [n][k] layout, packed b32 stores
    for (int i = tid; i < 32 * 16; i += 256) {
      int nn = i >> 4, kp = (i & 15) << 1;
      int n = n0 + nn;
      v2h dv, mv;
      dv[0] = (_Float16)0.f; dv[1] = (_Float16)0.f;
      mv[0] = (_Float16)0.f; mv[1] = (_Float16)0.f;
      if (n < Ntot) {
        int b = n / HW, p = n - b * HW;
        int y = p / Wd, x = p - y * Wd;
#pragma unroll
        for (int j = 0; j < 2; ++j) {
          int k = k0 + kp + j;
          if (k < Ktot) {
            int ic = k / KK, r = k - ic * KK;
            int ky = r / KS, kx = r - ky * KS;
            int yy = y + ky - PAD, xx = x + kx - PAD;
            if (yy >= 0 && yy < H && xx >= 0 && xx < Wd) {
              int idx = ((b * IC + ic) * H + yy) * Wd + xx;
              dv[j] = dh[idx];
              if (NEED_UM) mv[j] = vmh[idx];
            }
          }
        }
      }
      *(v2h*)(Bb + nn * LROW + kp) = dv;
      if (NEED_UM) *(v2h*)(B2b + nn * LROW + kp) = mv;
    }
  };

  // fragment coordinates (CDNA5 16-bit layout: lanes 0-15 take K {0-7,16-23},
  // lanes 16-31 take K {8-15,24-31})
  const int mrow = (mi << 4) + (lane & 15);
  const int ncol = (ni << 4) + (lane & 15);
  const int klo = (lane < 16) ? 0 : 8;
  const int khi = klo + 16;

  stage(0, 0);
  int cur = 0;
  for (int k0 = 0; k0 < Ktot; k0 += 32) {
    // drain our async copy, then one workgroup barrier per chunk
    asm volatile("s_wait_asynccnt 0x0" ::: "memory");
    __syncthreads();
    if (k0 + 32 < Ktot) stage(k0 + 32, cur ^ 1);  // overlap with compute below

    const _Float16* Ab  = As[cur];
    const _Float16* Bb  = Bs[cur];
    v8h alo = *(const v8h*)(Ab + mrow * LROW + klo);
    v8h ahi = *(const v8h*)(Ab + mrow * LROW + khi);
    v8h blo = *(const v8h*)(Bb + ncol * LROW + klo);
    v8h bhi = *(const v8h*)(Bb + ncol * LROW + khi);
    v16h af = __builtin_shufflevector(alo, ahi, 0, 1, 2, 3, 4, 5, 6, 7, 8, 9,
                                      10, 11, 12, 13, 14, 15);
    v16h bD = __builtin_shufflevector(blo, bhi, 0, 1, 2, 3, 4, 5, 6, 7, 8, 9,
                                      10, 11, 12, 13, 14, 15);
    accD = __builtin_amdgcn_wmma_f32_16x16x32_f16(false, af, false, bD,
                                                  (short)0, accD, false, false);
    if (NEED_UM) {
      const _Float16* B2b = Bs2[cur];
      v8h clo = *(const v8h*)(B2b + ncol * LROW + klo);
      v8h chi = *(const v8h*)(B2b + ncol * LROW + khi);
      v16h bM = __builtin_shufflevector(clo, chi, 0, 1, 2, 3, 4, 5, 6, 7, 8, 9,
                                        10, 11, 12, 13, 14, 15);
      // identical instruction as previous, same A operand -> reuse-A hint legal
      accM = __builtin_amdgcn_wmma_f32_16x16x32_f16(false, af, false, bM,
                                                    (short)0, accM, true, false);
    }
    cur ^= 1;
  }

  // ---- epilogue: D layout -> VGPR r holds M=r (+8 for lanes 16-31), N=lane%16
  const int mbase = m0 + (mi << 4) + ((lane >= 16) ? 8 : 0);
  const int n = n0 + (ni << 4) + (lane & 15);
  if (n < Ntot) {
    int b = n / HW;
    int p = n - b * HW;
#pragma unroll
    for (int r = 0; r < 8; ++r) {
      int m = mbase + r;
      if (m < OC) {
        int o = (b * OC + m) * HW + p;
        udiff[o] = accD[r];
        if (NEED_UM) um_out[o] = accM[r];
      }
    }
  }
}

// ------------------------------------------------------- NOSO neuron (plain)
__global__ void k_neuron(const float* __restrict__ udiff, float* __restrict__ sav,
                         float* __restrict__ sp, float* __restrict__ sumslot, int n) {
  int i = blockIdx.x * 256 + threadIdx.x;
  float s = 0.f;
  if (i < n) {
    float sv = sav[i];
    float u = sv * udiff[i];
    s = (u - 1.0f > 0.f) ? 1.f : 0.f;
    sav[i] = sv * (1.f - s);
    sp[i] = s;
  }
  blockAtomicSum(s, sumslot);
}

// -------------------------------------- NOSO neuron + pool latency bookkeeping
__global__ void k_neuron_pool(const float* __restrict__ udiff,
                              const float* __restrict__ um, float* __restrict__ sav,
                              float* __restrict__ sp, float* __restrict__ pe,
                              float* __restrict__ pl, float* __restrict__ sumslot,
                              int n) {
  int i = blockIdx.x * 256 + threadIdx.x;
  float s = 0.f;
  if (i < n) {
    float sv = sav[i];
    float u = sv * udiff[i];
    s = (u - 1.0f > 0.f) ? 1.f : 0.f;
    sav[i] = sv * (1.f - s);
    sp[i] = s;
    float pev = pe[i];
    float nz = (um[i] != 0.f) ? 1.f : 0.f;
    float pen = pev + (((pev + nz) != 0.f) ? 1.f : 0.f);
    pe[i] = pen;
    pl[i] += s * (pen - NSTEPS + 1.0f);
  }
  blockAtomicSum(s, sumslot);
}

// ------------------------------------------------ output neuron (f3 + readout)
__global__ void k_neuron_out(const float* __restrict__ udiff, float* __restrict__ sav,
                             float* __restrict__ sp, float* __restrict__ te,
                             float* __restrict__ out_t, float* __restrict__ out_u,
                             float* __restrict__ sumslot, int n) {
  int i = blockIdx.x * 256 + threadIdx.x;
  float s = 0.f;
  if (i < n) {
    float sv = sav[i];
    float u = sv * udiff[i];
    s = (u - 1.0f > 0.f) ? 1.f : 0.f;
    sav[i] = sv * (1.f - s);
    sp[i] = s;
    float tev = te[i];
    float nz = (u != 0.f) ? 1.f : 0.f;
    float ten = tev + (((tev + nz) != 0.f) ? 1.f : 0.f);
    te[i] = ten;
    out_t[i] += s * (ten - NSTEPS);
    out_u[i] += s * u;
  }
  blockAtomicSum(s, sumslot);
}

// ------------------------------------------------------- latency 2x2 min-pool
__global__ void k_minpool(const float* __restrict__ sp, const float* __restrict__ lat,
                          float* __restrict__ out, int B, int C, int H, int W) {
  const int OH = H >> 1, OW = W >> 1;
  const int n = B * C * OH * OW;
  int i = blockIdx.x * 256 + threadIdx.x;
  if (i >= n) return;
  int ox = i % OW;
  int t = i / OW;
  int oy = t % OH;
  t /= OH;
  int c = t % C;
  int b = t / C;
  int base = ((b * C + c) * H + oy * 2) * W + ox * 2;
  float bestl = lat[base];
  float bests = sp[base];
#pragma unroll
  for (int k = 1; k < 4; ++k) {
    int idx = base + (k >> 1) * W + (k & 1);
    float l = lat[idx];
    if (l < bestl) { bestl = l; bests = sp[idx]; }  // strict < == first-min argmin
  }
  out[i] = bests;
}

// ------------------------------------------------------------- output gather
__global__ void k_gather(const float* __restrict__ out_t, const float* __restrict__ out_u,
                         const float* __restrict__ sum_sp, float* __restrict__ out) {
  int i = threadIdx.x;
  if (i < 80) { out[i] = out_t[i]; out[80 + i] = out_u[i]; }
  if (i < 9) out[160 + i] = sum_sp[i];
}

// ===========================================================================
extern "C" void kernel_launch(void* const* d_in, const int* in_sizes, int n_in,
                              void* d_out, int out_size, void* d_ws, size_t ws_size,
                              hipStream_t stream) {
  (void)in_sizes; (void)n_in; (void)out_size; (void)ws_size;
  const float* input = (const float*)d_in[0];
  const float* wc1 = (const float*)d_in[1];
  const float* wc2 = (const float*)d_in[2];
  const float* wc3 = (const float*)d_in[3];
  const float* wc4 = (const float*)d_in[4];
  const float* wc5 = (const float*)d_in[5];
  const float* wf1 = (const float*)d_in[6];
  const float* wf2 = (const float*)d_in[7];
  const float* wf3 = (const float*)d_in[8];
  float* out = (float*)d_out;

  const int B = 8;
  float* ws = (float*)d_ws;
  size_t off = 0;
  auto alloc = [&](size_t nf) { float* p = ws + off; off += nf; return p; };
  auto alloch = [&](size_t nh) {
    size_t nf = ((nh + 1) / 2 + 3) & ~(size_t)3;  // keep 16B alignment
    return (_Float16*)alloc(nf);
  };

  // --- f32 state (NCHW, reference layout)
  float* in_u = alloc(B * 3 * 1024);
  float* in_sp = alloc(B * 3 * 1024);
  float* sav1 = alloc(B * 64 * 1024);
  float* vm1 = alloc(B * 3 * 1024);   float* vs1 = alloc(B * 3 * 1024);
  float* sp1 = alloc(B * 64 * 1024);
  float* sav2 = alloc(B * 128 * 1024);
  float* vm2 = alloc(B * 64 * 1024);  float* vs2 = alloc(B * 64 * 1024);
  float* sp2 = alloc(B * 128 * 1024);
  float* p1e = alloc(B * 128 * 1024); float* p1l = alloc(B * 128 * 1024);
  float* sp2p = alloc(B * 128 * 256);
  float* sav3 = alloc(B * 256 * 256);
  float* vm3 = alloc(B * 128 * 256);  float* vs3 = alloc(B * 128 * 256);
  float* sp3 = alloc(B * 256 * 256);
  float* p2e = alloc(B * 256 * 256);  float* p2l = alloc(B * 256 * 256);
  float* sp3p = alloc(B * 256 * 64);
  float* sav4 = alloc(B * 512 * 64);
  float* vm4 = alloc(B * 256 * 64);   float* vs4 = alloc(B * 256 * 64);
  float* sp4 = alloc(B * 512 * 64);
  float* sav5 = alloc(B * 256 * 64);
  float* vm5 = alloc(B * 512 * 64);   float* vs5 = alloc(B * 512 * 64);
  float* sp5 = alloc(B * 256 * 64);   // == fc1 input (B x 16384)
  float* savf1 = alloc(B * 1024);
  float* vmf1 = alloc(B * 16384);     float* vsf1 = alloc(B * 16384);
  float* spf1 = alloc(B * 1024);
  float* savf2 = alloc(B * 512);
  float* vmf2 = alloc(B * 1024);      float* vsf2 = alloc(B * 1024);
  float* spf2 = alloc(B * 512);
  float* savf3 = alloc(B * 16);
  float* vmf3 = alloc(B * 512);       float* vsf3 = alloc(B * 512);
  float* spf3 = alloc(B * 16);
  float* te = alloc(B * 16);
  float* out_t = alloc(B * 16);
  float* out_u = alloc(B * 16);
  float* sum_sp = alloc(16);
  float* udiff = alloc(B * 128 * 1024);  // largest GEMM output (conv2)
  float* umbuf = alloc(B * 128 * 1024);

  // --- f16 weights (converted once, reused across all 8 steps)
  _Float16* wc1h = alloch(64 * 75);
  _Float16* wc2h = alloch(128 * 1600);
  _Float16* wc3h = alloch(256 * 3200);
  _Float16* wc4h = alloch(512 * 6400);
  _Float16* wc5h = alloch(256 * 12800);
  _Float16* wf1h = alloch(1024 * 16384);
  _Float16* wf2h = alloch(512 * 1024);
  _Float16* wf3h = alloch(16 * 512);
  // --- f16 trace buffers per layer input: dh = vm - vs, mh = vm
  _Float16* d1h = alloch(B * 3 * 1024);   _Float16* m1h = alloch(B * 3 * 1024);
  _Float16* d2h = alloch(B * 64 * 1024);  _Float16* m2h = alloch(B * 64 * 1024);
  _Float16* d3h = alloch(B * 128 * 256);  _Float16* m3h = alloch(B * 128 * 256);
  _Float16* d4h = alloch(B * 256 * 64);   _Float16* m4h = alloch(B * 256 * 64);
  _Float16* d5h = alloch(B * 512 * 64);   _Float16* m5h = alloch(B * 512 * 64);
  _Float16* df1h = alloch(B * 16384);     _Float16* mf1h = alloch(B * 16384);
  _Float16* df2h = alloch(B * 1024);      _Float16* mf2h = alloch(B * 1024);
  _Float16* df3h = alloch(B * 512);       _Float16* mf3h = alloch(B * 512);

  auto fill = [&](float* p, int n, float v) {
    k_fill<<<TDIV(n, 256), 256, 0, stream>>>(p, n, v);
  };
  auto cvt = [&](const float* s, _Float16* d, int n) {
    k_f32tof16<<<TDIV(n, 256), 256, 0, stream>>>(s, d, n);
  };
  // weight conversion (once per launch; deterministic)
  cvt(wc1, wc1h, 64 * 75);     cvt(wc2, wc2h, 128 * 1600);
  cvt(wc3, wc3h, 256 * 3200);  cvt(wc4, wc4h, 512 * 6400);
  cvt(wc5, wc5h, 256 * 12800); cvt(wf1, wf1h, 1024 * 16384);
  cvt(wf2, wf2h, 512 * 1024);  cvt(wf3, wf3h, 10 * 512);

  // --- initialize state
  fill(in_u, B * 3 * 1024, 0.f);
  fill(sav1, B * 64 * 1024, 1.f);  fill(vm1, B * 3 * 1024, 0.f);  fill(vs1, B * 3 * 1024, 0.f);
  fill(sav2, B * 128 * 1024, 1.f); fill(vm2, B * 64 * 1024, 0.f); fill(vs2, B * 64 * 1024, 0.f);
  fill(sav3, B * 256 * 256, 1.f);  fill(vm3, B * 128 * 256, 0.f); fill(vs3, B * 128 * 256, 0.f);
  fill(sav4, B * 512 * 64, 1.f);   fill(vm4, B * 256 * 64, 0.f);  fill(vs4, B * 256 * 64, 0.f);
  fill(sav5, B * 256 * 64, 1.f);   fill(vm5, B * 512 * 64, 0.f);  fill(vs5, B * 512 * 64, 0.f);
  fill(savf1, B * 1024, 1.f); fill(vmf1, B * 16384, 0.f); fill(vsf1, B * 16384, 0.f);
  fill(savf2, B * 512, 1.f);  fill(vmf2, B * 1024, 0.f);  fill(vsf2, B * 1024, 0.f);
  fill(savf3, B * 10, 1.f);   fill(vmf3, B * 512, 0.f);   fill(vsf3, B * 512, 0.f);
  fill(p1e, B * 128 * 1024, 0.f); fill(p1l, B * 128 * 1024, NSTEPS);
  fill(p2e, B * 256 * 256, 0.f);  fill(p2l, B * 256 * 256, NSTEPS);
  fill(te, B * 10, 0.f); fill(out_t, B * 10, NSTEPS); fill(out_u, B * 10, 0.f);
  fill(sum_sp, 16, 0.f);

  auto gemm = [&](const _Float16* Wt, const _Float16* dh, const _Float16* mh,
                  float* ud, float* um, int OC, int IC, int KS, int PAD, int H,
                  int Wd, bool need_um) {
    dim3 grid(TDIV(B * H * Wd, 32), TDIV(OC, 64));
    if (need_um)
      k_noso_gemm<true><<<grid, 256, 0, stream>>>(Wt, dh, mh, ud, um, OC, IC, KS, PAD, H, Wd, B);
    else
      k_noso_gemm<false><<<grid, 256, 0, stream>>>(Wt, dh, mh, ud, nullptr, OC, IC, KS, PAD, H, Wd, B);
  };
  auto trace = [&](const float* sp, float* vm, float* vs, _Float16* dh,
                   _Float16* mh, int n) {
    k_trace<<<TDIV(n, 256), 256, 0, stream>>>(sp, vm, vs, dh, mh, n);
  };
  auto neuron = [&](const float* ud, float* sav, float* sp, int slot, int n) {
    k_neuron<<<TDIV(n, 256), 256, 0, stream>>>(ud, sav, sp, sum_sp + slot, n);
  };

  // ======================= 8 time steps =======================
  for (int t = 0; t < 8; ++t) {
    k_input_encode<<<TDIV(B * 3 * 1024, 256), 256, 0, stream>>>(
        input, in_u, in_sp, sum_sp + 0, B * 3 * 1024);
    // conv1: 3 -> 64 @ 32x32
    trace(in_sp, vm1, vs1, d1h, m1h, B * 3 * 1024);
    gemm(wc1h, d1h, m1h, udiff, nullptr, 64, 3, 5, 2, 32, 32, false);
    neuron(udiff, sav1, sp1, 1, B * 64 * 1024);
    // conv2: 64 -> 128 @ 32x32 (needs um for pool latency)
    trace(sp1, vm2, vs2, d2h, m2h, B * 64 * 1024);
    gemm(wc2h, d2h, m2h, udiff, umbuf, 128, 64, 5, 2, 32, 32, true);
    k_neuron_pool<<<TDIV(B * 128 * 1024, 256), 256, 0, stream>>>(
        udiff, umbuf, sav2, sp2, p1e, p1l, sum_sp + 2, B * 128 * 1024);
    k_minpool<<<TDIV(B * 128 * 256, 256), 256, 0, stream>>>(sp2, p1l, sp2p, B, 128, 32, 32);
    // conv3: 128 -> 256 @ 16x16 (needs um)
    trace(sp2p, vm3, vs3, d3h, m3h, B * 128 * 256);
    gemm(wc3h, d3h, m3h, udiff, umbuf, 256, 128, 5, 2, 16, 16, true);
    k_neuron_pool<<<TDIV(B * 256 * 256, 256), 256, 0, stream>>>(
        udiff, umbuf, sav3, sp3, p2e, p2l, sum_sp + 3, B * 256 * 256);
    k_minpool<<<TDIV(B * 256 * 64, 256), 256, 0, stream>>>(sp3, p2l, sp3p, B, 256, 16, 16);
    // conv4: 256 -> 512 @ 8x8
    trace(sp3p, vm4, vs4, d4h, m4h, B * 256 * 64);
    gemm(wc4h, d4h, m4h, udiff, nullptr, 512, 256, 5, 2, 8, 8, false);
    neuron(udiff, sav4, sp4, 4, B * 512 * 64);
    // conv5: 512 -> 256 @ 8x8
    trace(sp4, vm5, vs5, d5h, m5h, B * 512 * 64);
    gemm(wc5h, d5h, m5h, udiff, nullptr, 256, 512, 5, 2, 8, 8, false);
    neuron(udiff, sav5, sp5, 5, B * 256 * 64);
    // fc1: 16384 -> 1024
    trace(sp5, vmf1, vsf1, df1h, mf1h, B * 16384);
    gemm(wf1h, df1h, mf1h, udiff, nullptr, 1024, 16384, 1, 0, 1, 1, false);
    neuron(udiff, savf1, spf1, 6, B * 1024);
    // fc2: 1024 -> 512
    trace(spf1, vmf2, vsf2, df2h, mf2h, B * 1024);
    gemm(wf2h, df2h, mf2h, udiff, nullptr, 512, 1024, 1, 0, 1, 1, false);
    neuron(udiff, savf2, spf2, 7, B * 512);
    // fc3: 512 -> 10 + temporal readout
    trace(spf2, vmf3, vsf3, df3h, mf3h, B * 512);
    gemm(wf3h, df3h, mf3h, udiff, nullptr, 10, 512, 1, 0, 1, 1, false);
    k_neuron_out<<<1, 256, 0, stream>>>(udiff, savf3, spf3, te, out_t, out_u,
                                        sum_sp + 8, B * 10);
  }

  k_gather<<<1, 128, 0, stream>>>(out_t, out_u, sum_sp, out);
}
